// SlidingWindowAttention_7043746365913
// MI455X (gfx1250) — matrix operand order repaired and checked by reading, not compiled, not run on hardware
//
#include <hip/hip_runtime.h>

typedef __attribute__((ext_vector_type(16))) _Float16 v16h;
typedef __attribute__((ext_vector_type(4)))  _Float16 h4;
typedef __attribute__((ext_vector_type(2)))  _Float16 h2;
typedef __attribute__((ext_vector_type(8)))  float    v8f;
typedef __attribute__((ext_vector_type(4)))  float    f4;
typedef __attribute__((ext_vector_type(4)))  uint32_t u32x4;
typedef __attribute__((ext_vector_type(8)))  uint32_t u32x8;

#define WMMA_F16(A, B, C) \
  __builtin_amdgcn_wmma_f32_16x16x32_f16(false, (A), false, (B), (short)0, (C), false, false)

// ---------------------------------------------------------------------------
// TDM: async DMA of a 2D f16 tile (tile_d0 x tile_d1, row stride stride0)
// from global memory into LDS, padding each 32-half row to 40 halves
// (pad_interval = 16 dwords, pad_amount = 4 dwords) to match the LDS layout.
// Tracked with TENSORcnt; caller must s_wait_tensorcnt before consuming.
// ---------------------------------------------------------------------------
__device__ __forceinline__ void tdm_load_tile_2d(uint32_t lds_addr,
                                                 const _Float16* gptr,
                                                 uint32_t tensor_d0, uint32_t tensor_d1,
                                                 uint32_t tile_d0,   uint32_t tile_d1,
                                                 uint32_t stride0)
{
    uint64_t ga = (uint64_t)(uintptr_t)gptr;
    u32x4 g0;
    g0[0] = 1u;                                     // count=1, user descriptor
    g0[1] = lds_addr;                               // LDS byte address
    g0[2] = (uint32_t)ga;                           // global_addr[31:0]
    g0[3] = (uint32_t)(ga >> 32) | (2u << 30);      // global_addr[56:32] | type=2
    u32x8 g1;
    g1[0] = (1u << 16)                              // data_size = 2 bytes
          | (1u << 20)                              // pad_enable
          | (3u << 22)                              // pad_interval: 16 dwords
          | (3u << 25);                             // pad_amount:   4 dwords
    g1[1] = (tensor_d0 & 0xFFFFu) << 16;            // tensor_dim0[15:0]
    g1[2] = (tensor_d0 >> 16) | ((tensor_d1 & 0xFFFFu) << 16);
    g1[3] = (tensor_d1 >> 16) | (tile_d0 << 16);    // tile_dim0
    g1[4] = tile_d1;                                // tile_dim1 (tile_dim2 = 0)
    g1[5] = stride0;                                // tensor_dim0_stride[31:0]
    g1[6] = 0u;
    g1[7] = 0u;
    asm volatile("tensor_load_to_lds %0, %1" :: "s"(g0), "s"(g1) : "memory");
}

// ---------------------------------------------------------------------------
// Elementwise f32 -> f16 conversion (vectorized x4).
// ---------------------------------------------------------------------------
__global__ __launch_bounds__(256)
void cvt_f16_kernel(const float* __restrict__ src, _Float16* __restrict__ dst)
{
    int i = (blockIdx.x * 256 + threadIdx.x) * 4;
    f4 v = *(const f4*)(src + i);
    h4 h = { (_Float16)v.x, (_Float16)v.y, (_Float16)v.z, (_Float16)v.w };
    *(h4*)(dst + i) = h;
}

// ---------------------------------------------------------------------------
// Weight convert + transpose: Wt[n][k] = (f16)W[k][n].  32x32 LDS tiles.
// ---------------------------------------------------------------------------
__global__ __launch_bounds__(256)
void wtrans_kernel(const float* __restrict__ W, _Float16* __restrict__ Wt, int K, int N)
{
    __shared__ float tile[32][33];
    const int n0 = blockIdx.x * 32;
    const int k0 = blockIdx.y * 32;
    const int tx = threadIdx.x & 31;
    const int ty = threadIdx.x >> 5;
    #pragma unroll
    for (int r = 0; r < 4; r++)
        tile[ty + 8 * r][tx] = W[(size_t)(k0 + ty + 8 * r) * N + n0 + tx];
    __syncthreads();
    #pragma unroll
    for (int r = 0; r < 4; r++)
        Wt[(size_t)(n0 + ty + 8 * r) * K + k0 + tx] = (_Float16)tile[tx][ty + 8 * r];
}

// ---------------------------------------------------------------------------
// Tiled GEMM with TDM-staged, double-buffered LDS tiles (all-f16 operands):
//   C[M x N] = A[M x K] * Wt[N x K]^T + bias
//   128 threads = 4 waves; 64x64 tile; BK=32; wave 0 drives the TDM.
// ---------------------------------------------------------------------------
template<bool OUT_F32>
__global__ __launch_bounds__(128)
void gemm64_tdm_kernel(const _Float16* __restrict__ A,    // [M][K]
                       const _Float16* __restrict__ Wt,   // [N][K]
                       const float* __restrict__ bias,
                       _Float16* __restrict__ Cf16,
                       float* __restrict__ Cf32,
                       int M, int K, int N)
{
    __shared__ _Float16 Ah[2][64][40];   // [m][k], TDM-padded rows
    __shared__ _Float16 Bt[2][64][40];   // [n][k], TDM-padded rows

    const int bm = blockIdx.x;
    const int bn = blockIdx.y;
    const int tid  = threadIdx.x;
    const int wave = tid >> 5;
    const int lane = tid & 31;
    const int hi   = lane >> 4;
    const int lm   = lane & 15;
    const int wm   = (wave >> 1) * 32;
    const int wn   = (wave & 1)  * 32;
    const int kbA  = hi * 16;
    const int kbB  = hi * 16;
    const bool issuer = (wave == 0);
    const int  nkt = K / 32;

    const _Float16* Abase = A  + (size_t)(bm * 64) * K;
    const _Float16* Wbase = Wt + (size_t)(bn * 64) * K;

    if (issuer) {
        tdm_load_tile_2d((uint32_t)(uintptr_t)&Ah[0][0][0], Abase, K, M, 32, 64, K);
        tdm_load_tile_2d((uint32_t)(uintptr_t)&Bt[0][0][0], Wbase, K, N, 32, 64, K);
    }

    v8f acc[2][2] = {};

    for (int kt = 0; kt < nkt; kt++) {
        if (issuer) __builtin_amdgcn_s_wait_tensorcnt(0);
        __syncthreads();
        if (issuer && kt + 1 < nkt) {
            int nb = (kt + 1) & 1;
            int k0 = (kt + 1) * 32;
            tdm_load_tile_2d((uint32_t)(uintptr_t)&Ah[nb][0][0], Abase + k0, K, M, 32, 64, K);
            tdm_load_tile_2d((uint32_t)(uintptr_t)&Bt[nb][0][0], Wbase + k0, K, N, 32, 64, K);
        }
        const int buf = kt & 1;

        // ---- fragments: A row / B col = lm, contiguous k = 16*hi + s in both ----
        v16h a[2], b[2];
        #pragma unroll
        for (int t = 0; t < 2; t++) {
            const _Float16* arow = &Ah[buf][wm + t * 16 + lm][0];
            h2* av = (h2*)&a[t];
            #pragma unroll
            for (int i = 0; i < 4; i++) av[i]     = *(const h2*)(arow + kbA + 2 * i);
            #pragma unroll
            for (int i = 0; i < 4; i++) av[4 + i] = *(const h2*)(arow + 8 + kbA + 2 * i);

            const _Float16* brow = &Bt[buf][wn + t * 16 + lm][0];
            h2* bv = (h2*)&b[t];
            #pragma unroll
            for (int i = 0; i < 8; i++) bv[i] = *(const h2*)(brow + kbB + 2 * i);
        }

        acc[0][0] = WMMA_F16(a[0], b[0], acc[0][0]);
        acc[0][1] = WMMA_F16(a[0], b[1], acc[0][1]);
        acc[1][0] = WMMA_F16(a[1], b[0], acc[1][0]);
        acc[1][1] = WMMA_F16(a[1], b[1], acc[1][1]);
    }

    // ---- epilogue: bias + store (C/D layout: row = i + 8*hi, col = lm) ----
    #pragma unroll
    for (int tm = 0; tm < 2; tm++) {
        #pragma unroll
        for (int tn = 0; tn < 2; tn++) {
            int col = bn * 64 + wn + tn * 16 + lm;
            float bval = bias[col];
            #pragma unroll
            for (int i = 0; i < 8; i++) {
                int row = bm * 64 + wm + tm * 16 + i + 8 * hi;
                float v = acc[tm][tn][i] + bval;
                if constexpr (OUT_F32) Cf32[(size_t)row * N + col] = v;
                else                   Cf16[(size_t)row * N + col] = (_Float16)v;
            }
        }
    }
}

// ---------------------------------------------------------------------------
// Sliding-window attention core. One wave per (16-query tile, head).
//   S=2048, D=768, H=12, DH=64, window = +/-64 (129 wide).
//   Key span per q-tile: [s0-64, s0+79] = 144 keys = 9 tiles of 16,
//   padded to 160 for K=32 WMMA chunks in the PV matmul.
// ---------------------------------------------------------------------------
__global__ __launch_bounds__(32)
void swattn_kernel(const _Float16* __restrict__ Q,
                   const _Float16* __restrict__ Km,
                   const _Float16* __restrict__ Vm,
                   _Float16* __restrict__ Ctx)
{
    constexpr int S = 2048, D = 768, DH = 64;
    __shared__ _Float16 Plds[16][168];   // probs, K-padded to 160
    __shared__ _Float16 Vt[64][168];     // V window transposed: [dh][k]

    const int s0   = blockIdx.x * 16;
    const int h    = blockIdx.y;
    const int lane = threadIdx.x;
    const int hi   = lane >> 4;
    const int lm   = lane & 15;
    const int kbA  = hi * 16;
    const int kbB  = hi * 16;
    const int kb0  = s0 - 64;            // first key of the window span

    // ---- Q fragments (16 x 64 over two K=32 chunks), A-matrix layout ----
    const _Float16* qrow = Q + (size_t)(s0 + lm) * D + h * DH;
    v16h aq[2];
    #pragma unroll
    for (int c = 0; c < 2; c++) {
        h2* av = (h2*)&aq[c];
        #pragma unroll
        for (int i = 0; i < 4; i++) av[i]     = *(const h2*)(qrow + c * 32 + kbA + 2 * i);
        #pragma unroll
        for (int i = 0; i < 4; i++) av[4 + i] = *(const h2*)(qrow + c * 32 + 8 + kbA + 2 * i);
    }

    // ---- scores: 9 key tiles, 2 WMMAs each (K-dim 64) ----
    float p[9][8];
    #pragma unroll
    for (int j = 0; j < 9; j++) {
        int key  = kb0 + j * 16 + lm;                  // this lane's key column
        int keyc = min(max(key, 0), S - 1);
        const _Float16* krow = Km + (size_t)keyc * D + h * DH;
        v16h b0, b1;
        h2* bv0 = (h2*)&b0;
        h2* bv1 = (h2*)&b1;
        #pragma unroll
        for (int i = 0; i < 8; i++) bv0[i] = *(const h2*)(krow + kbB + 2 * i);
        #pragma unroll
        for (int i = 0; i < 8; i++) bv1[i] = *(const h2*)(krow + 32 + kbB + 2 * i);
        v8f c = {};
        c = WMMA_F16(aq[0], b0, c);
        c = WMMA_F16(aq[1], b1, c);
        #pragma unroll
        for (int i = 0; i < 8; i++) p[j][i] = c[i];
    }

    // ---- mask + scale (C layout: row = i + 8*hi, col key = kb0+j*16+lm) ----
    const float scale = 0.125f;                        // 1/sqrt(64)
    float mx[8];
    #pragma unroll
    for (int i = 0; i < 8; i++) mx[i] = -INFINITY;
    #pragma unroll
    for (int j = 0; j < 9; j++) {
        int key = kb0 + j * 16 + lm;
        #pragma unroll
        for (int i = 0; i < 8; i++) {
            int s = s0 + i + 8 * hi;
            bool ok = (key >= 0) & (key < S) & (key >= s - 64) & (key <= s + 64);
            float v = ok ? p[j][i] * scale : -INFINITY;
            p[j][i] = v;
            mx[i] = fmaxf(mx[i], v);
        }
    }
    // row reductions stay inside each 16-lane half (rows live across 16 lanes)
    #pragma unroll
    for (int i = 0; i < 8; i++) {
        #pragma unroll
        for (int m = 1; m <= 8; m <<= 1)
            mx[i] = fmaxf(mx[i], __shfl_xor(mx[i], m, 32));
    }
    float sum[8] = {};
    #pragma unroll
    for (int j = 0; j < 9; j++)
        #pragma unroll
        for (int i = 0; i < 8; i++) {
            float e = __expf(p[j][i] - mx[i]);
            p[j][i] = e;
            sum[i] += e;
        }
    #pragma unroll
    for (int i = 0; i < 8; i++) {
        #pragma unroll
        for (int m = 1; m <= 8; m <<= 1)
            sum[i] += __shfl_xor(sum[i], m, 32);
        sum[i] = 1.0f / sum[i];
    }

    // ---- write P to LDS (f16), zero the K-pad 144..159 ----
    #pragma unroll
    for (int j = 0; j < 9; j++)
        #pragma unroll
        for (int i = 0; i < 8; i++)
            Plds[i + 8 * hi][j * 16 + lm] = (_Float16)(p[j][i] * sum[i]);
    #pragma unroll
    for (int i = 0; i < 8; i++)
        Plds[i + 8 * hi][144 + lm] = (_Float16)0.f;

    // ---- stage V window transposed into LDS: Vt[dh][k] ----
    for (int k = lane; k < 160; k += 32) {
        int key = kb0 + k;
        bool ok = (k < 144) & (key >= 0) & (key < S);
        const _Float16* vrow = Vm + (size_t)min(max(key, 0), S - 1) * D + h * DH;
        #pragma unroll
        for (int d = 0; d < 64; d++)
            Vt[d][k] = ok ? vrow[d] : (_Float16)0.f;
    }
    __syncthreads();

    // ---- ctx = P(16x160) x Vwin(160x64): 4 N-tiles x 5 K-chunks ----
    #pragma unroll
    for (int nt = 0; nt < 4; nt++) {
        v8f acc = {};
        #pragma unroll
        for (int c = 0; c < 5; c++) {
            v16h a, b;
            const _Float16* prow = &Plds[lm][c * 32];
            h2* av = (h2*)&a;
            #pragma unroll
            for (int i = 0; i < 4; i++) av[i]     = *(const h2*)(prow + kbA + 2 * i);
            #pragma unroll
            for (int i = 0; i < 4; i++) av[4 + i] = *(const h2*)(prow + 8 + kbA + 2 * i);
            const _Float16* vrow = &Vt[nt * 16 + lm][c * 32];
            h2* bv = (h2*)&b;
            #pragma unroll
            for (int i = 0; i < 8; i++) bv[i] = *(const h2*)(vrow + kbB + 2 * i);
            acc = WMMA_F16(a, b, acc);
        }
        #pragma unroll
        for (int i = 0; i < 8; i++)
            Ctx[(size_t)(s0 + i + 8 * hi) * D + h * DH + nt * 16 + lm] = (_Float16)acc[i];
    }
}

// ---------------------------------------------------------------------------
// Launcher: convert -> transpose weights -> QKV GEMMs -> attention -> out proj.
// ---------------------------------------------------------------------------
extern "C" void kernel_launch(void* const* d_in, const int* in_sizes, int n_in,
                              void* d_out, int out_size, void* d_ws, size_t ws_size,
                              hipStream_t stream) {
    const float* query = (const float*)d_in[0];
    const float* key   = (const float*)d_in[1];
    const float* value = (const float*)d_in[2];
    const float* Wq    = (const float*)d_in[3];
    const float* bq    = (const float*)d_in[4];
    const float* Wk    = (const float*)d_in[5];
    const float* bk    = (const float*)d_in[6];
    const float* Wv    = (const float*)d_in[7];
    const float* bv    = (const float*)d_in[8];
    const float* Wo    = (const float*)d_in[9];
    const float* bo    = (const float*)d_in[10];
    float* out = (float*)d_out;

    const size_t SD = (size_t)2048 * 768;   // activation elements
    const size_t DD = (size_t)768 * 768;    // weight elements
    _Float16* Xq  = (_Float16*)d_ws;        // f16 inputs
    _Float16* Xk  = Xq + SD;
    _Float16* Xv  = Xk + SD;
    _Float16* Qh  = Xv + SD;                // f16 projected Q/K/V, ctx
    _Float16* Kh  = Qh + SD;
    _Float16* Vh  = Kh + SD;
    _Float16* Ch  = Vh + SD;
    _Float16* Wqt = Ch + SD;                // f16 transposed weights [n][k]
    _Float16* Wkt = Wqt + DD;
    _Float16* Wvt = Wkt + DD;
    _Float16* Wot = Wvt + DD;

    // 1) activations -> f16
    dim3 cg((unsigned)(SD / 4 / 256)), cb(256);
    cvt_f16_kernel<<<cg, cb, 0, stream>>>(query, Xq);
    cvt_f16_kernel<<<cg, cb, 0, stream>>>(key,   Xk);
    cvt_f16_kernel<<<cg, cb, 0, stream>>>(value, Xv);

    // 2) weights -> f16, transposed to [n][k]
    dim3 tg(768 / 32, 768 / 32), tb(256);
    wtrans_kernel<<<tg, tb, 0, stream>>>(Wq, Wqt, 768, 768);
    wtrans_kernel<<<tg, tb, 0, stream>>>(Wk, Wkt, 768, 768);
    wtrans_kernel<<<tg, tb, 0, stream>>>(Wv, Wvt, 768, 768);
    wtrans_kernel<<<tg, tb, 0, stream>>>(Wo, Wot, 768, 768);

    // 3) QKV projections (TDM-staged WMMA GEMMs)
    dim3 gg(2048 / 64, 768 / 64), gb(128);
    gemm64_tdm_kernel<false><<<gg, gb, 0, stream>>>(Xq, Wqt, bq, Qh, nullptr, 2048, 768, 768);
    gemm64_tdm_kernel<false><<<gg, gb, 0, stream>>>(Xk, Wkt, bk, Kh, nullptr, 2048, 768, 768);
    gemm64_tdm_kernel<false><<<gg, gb, 0, stream>>>(Xv, Wvt, bv, Vh, nullptr, 2048, 768, 768);

    // 4) sliding-window attention
    swattn_kernel<<<dim3(2048 / 16, 12), dim3(32), 0, stream>>>(Qh, Kh, Vh, Ch);

    // 5) output projection -> fp32
    gemm64_tdm_kernel<true><<<gg, gb, 0, stream>>>(Ch, Wot, bo, nullptr, out, 2048, 768, 768);
}
